// model_79130477461574
// MI455X (gfx1250) — compile-verified, matrix operand order checked by reading, not run on hardware
//
#include <hip/hip_runtime.h>

// ---------------------------------------------------------------------------
// Model dims (match reference)
// ---------------------------------------------------------------------------
#define HSZd 1024
#define HHd   512      // per-direction hidden
#define GSZd 2048      // 4*HHd gate width (encoder)
#define BBd    32
#define SSd   512
#define TTd    16
#define NV  20000
#define VV  10000

typedef __attribute__((ext_vector_type(8)))  float  v8f;
typedef __attribute__((ext_vector_type(16))) __bf16 v16bf;
typedef __attribute__((ext_vector_type(8)))  __bf16 v8bf;

// ---------------------------------------------------------------------------
// Helpers
// ---------------------------------------------------------------------------
__device__ __forceinline__ float fsig(float x) {
    return 1.0f / (1.0f + __expf(-x));
}
__device__ __forceinline__ float ftanh(float x) {
    float e = __expf(2.0f * x);
    return (e - 1.0f) / (e + 1.0f);
}

// Async Global->LDS copy, 16B per lane (gfx1250 GLOBAL_LOAD_ASYNC_TO_LDS_B128,
// tracked by ASYNCcnt).  lds32 = low 32 bits of the generic shared-pointer
// (generic LDS addresses carry the LDS offset in [31:0], aperture in [63:32]).
__device__ __forceinline__ void async_copy_b128(unsigned lds32, const void* gaddr) {
    asm volatile("global_load_async_to_lds_b128 %0, %1, off"
                 :: "v"(lds32), "v"(gaddr) : "memory");
}
__device__ __forceinline__ void wait_async0() {
    asm volatile("s_wait_asynccnt 0" ::: "memory");
}

// Load one 16x32 bf16 WMMA operand fragment (A- or B-style) from a row-major
// tile. Per ISA 7.12.2: lanes 0-15 -> row = lane, K lo-half {0..7,16..23};
// lanes 16-31 -> same rows, K hi-half {8..15,24..31}. `stride` in elements,
// must be a multiple of 8; base must be 16B aligned.
__device__ __forceinline__ v16bf load_frag(const __bf16* base, int stride) {
    int lane = threadIdx.x & 31;
    int r    = lane & 15;
    int hf   = lane >> 4;
    const __bf16* p = base + r * stride + hf * 8;
    v8bf lo = *(const v8bf*)(p);
    v8bf hi = *(const v8bf*)(p + 16);
    v16bf o;
#pragma unroll
    for (int i = 0; i < 8; ++i) { o[i] = lo[i]; o[i + 8] = hi[i]; }
    return o;
}

// ---------------------------------------------------------------------------
// Generic WMMA GEMM:  C[M,N] = A[M,K](bf16) * B[N,K](f32, converted)^T + bias
// Block tile 128x128, 512 threads (16 waves in a 4x4 grid, 2x2 subtiles each),
// K-step 32.  A tile staged via async Global->LDS; B converted through VGPRs.
// ---------------------------------------------------------------------------
#define GBM 128
#define GBN 128
#define GBK 32
#define LDT 48   // LDS tile row stride (elements); 48*2=96B -> 16B-aligned rows

__global__ __launch_bounds__(512)
void gemm_bf16(const __bf16* __restrict__ A, const float* __restrict__ Bw,
               const float* __restrict__ bias0, const float* __restrict__ bias1,
               float* __restrict__ C, int M, int N, int K)
{
    __shared__ __bf16 As[GBM * LDT];
    __shared__ __bf16 Bs[GBN * LDT];

    const int m0   = blockIdx.x * GBM;
    const int n0   = blockIdx.y * GBN;
    const int tid  = threadIdx.x;
    const int wave = tid >> 5;
    const int lane = tid & 31;
    const int wr   = wave >> 2;   // 0..3 : covers m-subtiles {2wr, 2wr+1}
    const int wc   = wave & 3;    // 0..3 : covers n-subtiles {2wc, 2wc+1}

    v8f acc00 = {}; v8f acc01 = {}; v8f acc10 = {}; v8f acc11 = {};

    const int lrow = tid >> 2;        // 0..127
    const int lcol = (tid & 3) * 8;   // 0,8,16,24

    __bf16* aslot = As + lrow * LDT + lcol;
    const unsigned ldsA = (unsigned)(uintptr_t)aslot;

    for (int k0 = 0; k0 < K; k0 += GBK) {
        { // stage A (bf16) via async DMA to LDS
            int gm = m0 + lrow;
            if (gm < M) async_copy_b128(ldsA, A + (size_t)gm * K + k0 + lcol);
            else        *(v8bf*)aslot = (v8bf){};
        }
        { // stage B: fp32 -> bf16 through VGPRs
            int gn = n0 + lrow;
            v8bf bv = {};
            if (gn < N) {
                const float* bp = Bw + (size_t)gn * K + k0 + lcol;
                if (k0 + GBK < K)
                    __builtin_prefetch(bp + GBK, 0, 0);   // global_prefetch_b8
#pragma unroll
                for (int i = 0; i < 8; ++i) bv[i] = (__bf16)bp[i];
            }
            *(v8bf*)(Bs + lrow * LDT + lcol) = bv;
        }
        wait_async0();          // this wave's async tile bytes landed in LDS
        __syncthreads();        // whole tile visible to all waves

        v16bf a0 = load_frag(As + (wr * 32) * LDT, LDT);
        v16bf a1 = load_frag(As + (wr * 32 + 16) * LDT, LDT);
        v16bf b0 = load_frag(Bs + (wc * 32) * LDT, LDT);
        v16bf b1 = load_frag(Bs + (wc * 32 + 16) * LDT, LDT);
        acc00 = __builtin_amdgcn_wmma_f32_16x16x32_bf16(false, a0, false, b0,
                                                        (short)0, acc00, false, false);
        acc01 = __builtin_amdgcn_wmma_f32_16x16x32_bf16(false, a0, false, b1,
                                                        (short)0, acc01, false, false);
        acc10 = __builtin_amdgcn_wmma_f32_16x16x32_bf16(false, a1, false, b0,
                                                        (short)0, acc10, false, false);
        acc11 = __builtin_amdgcn_wmma_f32_16x16x32_bf16(false, a1, false, b1,
                                                        (short)0, acc11, false, false);
        __syncthreads();
    }

    const int r = lane & 15, hf = lane >> 4;
    v8f* accs[2][2] = { { &acc00, &acc01 }, { &acc10, &acc11 } };
#pragma unroll
    for (int i = 0; i < 2; ++i) {
#pragma unroll
        for (int j = 0; j < 2; ++j) {
            v8f a = *accs[i][j];
            int gnb = n0 + (wc * 2 + j) * 16 + r;
            if (gnb >= N) continue;
            float badd = 0.0f;
            if (bias0) badd += bias0[gnb];
            if (bias1) badd += bias1[gnb];
#pragma unroll
            for (int e = 0; e < 8; ++e) {
                int gm = m0 + (wr * 2 + i) * 16 + e + hf * 8;
                if (gm < M) C[(size_t)gm * N + gnb] = a[e] + badd;
            }
        }
    }
}

// ---------------------------------------------------------------------------
// Persistent recurrent LSTM scan.  Grid = 4 blocks: (dir, batch-half).
// The recurrence is independent per batch row, so batch splits freely.
// gates(t) = gpre[t] (precomputed x@Wih^T + biases) + h(t-1) @ Whh^T (WMMA).
// h lives in LDS (bf16, 16 rows); c lives in VGPRs: each of the 32 waves owns
// exactly one fixed 16x16 (batch,hidden) tile for the whole scan.
// ---------------------------------------------------------------------------
#define HST 536   // LDS h row stride (elements): mult of 8, rows 16B-aligned

__global__ __launch_bounds__(1024)
void lstm_scan(const float* __restrict__ gbase,    // [2][S][B][GSZ]
               const __bf16* __restrict__ whhbase, // [2][GSZ][HH] bf16
               __bf16* __restrict__ outBf,         // [S][B][HSZ] or null
               float*  __restrict__ outF,          // [B][S][HSZ] or null
               float*  __restrict__ hout,          // [B][HSZ] (this layer)
               float*  __restrict__ cout)          // [B][HSZ]
{
    __shared__ __bf16 hlds[16 * HST];

    const int dir   = blockIdx.x >> 1;
    const int mh    = blockIdx.x & 1;     // batch half
    const int bbase = mh * 16;
    const float*  g   = gbase   + (size_t)dir * SSd * BBd * GSZd;
    const __bf16* whh = whhbase + (size_t)dir * GSZd * HHd;

    const int tid = threadIdx.x, wave = tid >> 5, lane = tid & 31;
    const int r = lane & 15, hf = lane >> 4;
    const int jb = wave * 16;             // this wave's hidden column tile

    for (int i = tid; i < 16 * HST; i += 1024) hlds[i] = (__bf16)0.0f;
    __syncthreads();

    float cst[8];
    float hnew[8];
#pragma unroll
    for (int e = 0; e < 8; ++e) { cst[e] = 0.0f; hnew[e] = 0.0f; }

    for (int step = 0; step < SSd; ++step) {
        const int t = dir ? (SSd - 1 - step) : step;
        const float* gt = g + (size_t)t * BBd * GSZd;

        v8f acc[4];
#pragma unroll
        for (int ga = 0; ga < 4; ++ga) {
#pragma unroll
            for (int e = 0; e < 8; ++e) {
                int b = bbase + e + hf * 8;
                acc[ga][e] = gt[(size_t)b * GSZd + ga * HHd + jb + r];
            }
        }
        for (int k0 = 0; k0 < HHd; k0 += 32) {
            v16bf af = load_frag(hlds + k0, HST);
#pragma unroll
            for (int ga = 0; ga < 4; ++ga) {
                v16bf bf = load_frag(whh + (size_t)(ga * HHd + jb) * HHd + k0, HHd);
                acc[ga] = __builtin_amdgcn_wmma_f32_16x16x32_bf16(
                    false, af, false, bf, (short)0, acc[ga], false, false);
            }
        }
#pragma unroll
        for (int e = 0; e < 8; ++e) {
            float ig = fsig(acc[0][e]);
            float fg = fsig(acc[1][e]);
            float gg = ftanh(acc[2][e]);
            float og = fsig(acc[3][e]);
            float c2 = fg * cst[e] + ig * gg;
            cst[e]  = c2;
            hnew[e] = og * ftanh(c2);
        }
        __syncthreads();   // all reads of old h done
        const int j = jb + r;
#pragma unroll
        for (int e = 0; e < 8; ++e) {
            int bl = e + hf * 8;           // local batch row 0..15
            int b  = bbase + bl;
            float hv = hnew[e];
            hlds[bl * HST + j] = (__bf16)hv;
            if (outBf) outBf[((size_t)t * BBd + b) * HSZd + dir * HHd + j] = (__bf16)hv;
            else       outF [((size_t)b * SSd + t) * HSZd + dir * HHd + j] = hv;
        }
        __syncthreads();   // new h visible before next step reads
    }

    const int j = jb + r;
#pragma unroll
    for (int e = 0; e < 8; ++e) {
        int b = bbase + e + hf * 8;
        hout[(size_t)b * HSZd + dir * HHd + j] = hnew[e];
        cout[(size_t)b * HSZd + dir * HHd + j] = cst[e];
    }
}

// ---------------------------------------------------------------------------
// Small kernels
// ---------------------------------------------------------------------------
__global__ void f32_to_bf16(const float* __restrict__ src, __bf16* __restrict__ dst,
                            size_t n) {
    size_t i = (size_t)blockIdx.x * blockDim.x + threadIdx.x;
    if (i < n) dst[i] = (__bf16)src[i];
}

// x0[s][b][k] = bf16(encemb[inp[b][s]][k])
__global__ void embed_gather(const int* __restrict__ inp, const float* __restrict__ emb,
                             __bf16* __restrict__ x0) {
    size_t idx = (size_t)blockIdx.x * blockDim.x + threadIdx.x;
    if (idx >= (size_t)SSd * BBd * HSZd) return;
    int k = idx & (HSZd - 1);
    size_t sb = idx >> 10;
    int b = sb & (BBd - 1);
    int s = sb >> 5;
    int tok = inp[(size_t)b * SSd + s];
    x0[idx] = (__bf16)emb[(size_t)tok * HSZd + k];
}

// info[b][t][0:1024]=internoun[nouns[b][t]]; [1024:2048]=interverb[verbs[b][t]]
__global__ void info_gather(const int* __restrict__ nouns, const int* __restrict__ verbs,
                            const float* __restrict__ ne, const float* __restrict__ ve,
                            float* __restrict__ info) {
    size_t idx = (size_t)blockIdx.x * blockDim.x + threadIdx.x;
    if (idx >= (size_t)BBd * TTd * 2 * HSZd) return;
    int c = idx & (2 * HSZd - 1);
    size_t bt = idx >> 11;
    int t = bt & (TTd - 1);
    int b = bt >> 4;
    float v;
    if (c < HSZd) v = ne[(size_t)nouns[(size_t)b * TTd + t] * HSZd + c];
    else          v = ve[(size_t)verbs[(size_t)b * TTd + t] * HSZd + (c - HSZd)];
    info[idx] = v;
}

// inter LSTM gates: gout[b][n] = A[b]:W[n] (+ A2[b]:W2[n]) + b0[n] + b1[n]
__global__ void inter_gates(const float* __restrict__ A,  const float* __restrict__ W,
                            const float* __restrict__ A2, const float* __restrict__ W2,
                            const float* __restrict__ b0, const float* __restrict__ b1,
                            float* __restrict__ gout) {
    int idx = blockIdx.x * blockDim.x + threadIdx.x;   // 32 * 4096
    int b = idx >> 12, n = idx & 4095;
    const float* a = A + (size_t)b * HSZd;
    const float* w = W + (size_t)n * HSZd;
    float s = b0[n] + b1[n];
    for (int k = 0; k < HSZd; ++k) s += a[k] * w[k];
    if (A2) {
        const float* a2 = A2 + (size_t)b * HSZd;
        const float* w2 = W2 + (size_t)n * HSZd;
        for (int k = 0; k < HSZd; ++k) s += a2[k] * w2[k];
    }
    gout[idx] = s;
}

__global__ void inter_cell(const float* __restrict__ gates, const float* __restrict__ cin,
                           float* __restrict__ hout, __bf16* __restrict__ hbf) {
    int idx = blockIdx.x * blockDim.x + threadIdx.x;   // 32 * 1024
    int b = idx >> 10, j = idx & 1023;
    const float* gr = gates + (size_t)b * 4096;
    float ig = fsig(gr[j]);
    float fg = fsig(gr[1024 + j]);
    float gg = ftanh(gr[2048 + j]);
    float og = fsig(gr[3072 + j]);
    float c2 = fg * cin[idx] + ig * gg;
    float h  = og * ftanh(c2);
    hout[idx] = h;
    if (hbf) hbf[idx] = (__bf16)h;
}

// ---------------------------------------------------------------------------
// Launch
// ---------------------------------------------------------------------------
extern "C" void kernel_launch(void* const* d_in, const int* in_sizes, int n_in,
                              void* d_out, int out_size, void* d_ws, size_t ws_size,
                              hipStream_t stream) {
    (void)in_sizes; (void)n_in; (void)out_size; (void)ws_size;

    const int*   inp       = (const int*)d_in[0];
    const int*   verbs     = (const int*)d_in[1];
    const int*   nouns     = (const int*)d_in[2];
    const float* encemb    = (const float*)d_in[3];
    const float* internoun = (const float*)d_in[4];
    const float* interverb = (const float*)d_in[5];
    // enc[l].{fwd,bwd}.{Wih,Whh,bih,bhh}: 6 + (l*2+dir)*4 + {0..3}
    const float* encP[4][4];
    for (int dl = 0; dl < 4; ++dl)
        for (int i = 0; i < 4; ++i) encP[dl][i] = (const float*)d_in[6 + dl * 4 + i];
    const float* intP[2][4];
    for (int l = 0; l < 2; ++l)
        for (int i = 0; i < 4; ++i) intP[l][i] = (const float*)d_in[22 + l * 4 + i];
    const float* noungen_w = (const float*)d_in[30];
    const float* noungen_b = (const float*)d_in[31];
    const float* verbgen_w = (const float*)d_in[32];
    const float* verbgen_b = (const float*)d_in[33];

    // output regions (fp32, concatenated in return order)
    float* out  = (float*)d_out;
    float* enc  = out;
    float* nlog = enc  + (size_t)BBd * SSd * HSZd;
    float* vlog = nlog + (size_t)BBd * NV;
    float* info = vlog + (size_t)BBd * VV;

    // workspace layout
    const size_t SB = (size_t)SSd * BBd;            // 16384
    char* ws = (char*)d_ws;
    __bf16* x0bf   = (__bf16*)(ws);                                   // SB*1024 bf16
    __bf16* x1bf   = (__bf16*)(ws + 33554432);                        // SB*1024 bf16
    float*  gbuf   = (float*) (ws + 67108864);                        // 2*SB*2048 f32
    __bf16* whhbf  = (__bf16*)(ws + 67108864 + 268435456);            // 4*2048*512 bf16
    char*   ws2    =           ws + 67108864 + 268435456 + 8388608;
    float*  heBuf  = (float*)(ws2);                 // [2][32][1024]
    float*  ceBuf  = (float*)(ws2 + 262144);        // [2][32][1024]
    float*  h0buf  = (float*)(ws2 + 524288);        // [32][1024]
    float*  iout   = (float*)(ws2 + 655360);        // [32][1024]
    __bf16* ioutbf = (__bf16*)(ws2 + 786432);       // [32][1024] bf16
    float*  gatesb = (float*)(ws2 + 917504);        // [32][4096]

    // 1) one-time converts + gathers
    for (int dl = 0; dl < 4; ++dl) {
        size_t n = (size_t)GSZd * HHd;
        f32_to_bf16<<<(unsigned)((n + 255) / 256), 256, 0, stream>>>(
            encP[dl][1], whhbf + (size_t)dl * GSZd * HHd, n);
    }
    embed_gather<<<(unsigned)((SB * HSZd + 255) / 256), 256, 0, stream>>>(inp, encemb, x0bf);
    info_gather<<<(unsigned)(((size_t)BBd * TTd * 2 * HSZd + 255) / 256), 256, 0, stream>>>(
        nouns, verbs, internoun, interverb, info);

    // 2) layer 0: hoisted input GEMMs (both dirs), then recurrent scan
    dim3 gdim((unsigned)(SB / GBM), GSZd / GBN);
    gemm_bf16<<<gdim, 512, 0, stream>>>(x0bf, encP[0][0], encP[0][2], encP[0][3],
                                        gbuf, (int)SB, GSZd, HSZd);
    gemm_bf16<<<gdim, 512, 0, stream>>>(x0bf, encP[1][0], encP[1][2], encP[1][3],
                                        gbuf + SB * GSZd, (int)SB, GSZd, HSZd);
    lstm_scan<<<4, 1024, 0, stream>>>(gbuf, whhbf, x1bf, nullptr,
                                      heBuf, ceBuf);

    // 3) layer 1
    gemm_bf16<<<gdim, 512, 0, stream>>>(x1bf, encP[2][0], encP[2][2], encP[2][3],
                                        gbuf, (int)SB, GSZd, HSZd);
    gemm_bf16<<<gdim, 512, 0, stream>>>(x1bf, encP[3][0], encP[3][2], encP[3][3],
                                        gbuf + SB * GSZd, (int)SB, GSZd, HSZd);
    lstm_scan<<<4, 1024, 0, stream>>>(gbuf, whhbf + (size_t)2 * GSZd * HHd,
                                      nullptr, enc,
                                      heBuf + 32 * HSZd, ceBuf + 32 * HSZd);

    // 4) inter LSTM (1 step, 2 layers; layer-0 x-input is zero -> skip Wih term)
    inter_gates<<<(32 * 4096) / 256, 256, 0, stream>>>(
        heBuf, intP[0][1], nullptr, nullptr, intP[0][2], intP[0][3], gatesb);
    inter_cell<<<(32 * 1024) / 256, 256, 0, stream>>>(gatesb, ceBuf, h0buf, nullptr);
    inter_gates<<<(32 * 4096) / 256, 256, 0, stream>>>(
        h0buf, intP[1][0], heBuf + 32 * HSZd, intP[1][1], intP[1][2], intP[1][3], gatesb);
    inter_cell<<<(32 * 1024) / 256, 256, 0, stream>>>(gatesb, ceBuf + 32 * HSZd, iout, ioutbf);

    // 5) vocab logits via WMMA GEMM (weights stay fp32 in HBM; bf16-converted in tile)
    dim3 gn(1, (NV + GBN - 1) / GBN);
    gemm_bf16<<<gn, 512, 0, stream>>>(ioutbf, noungen_w, noungen_b, nullptr,
                                      nlog, BBd, NV, HSZd);
    dim3 gv(1, (VV + GBN - 1) / GBN);
    gemm_bf16<<<gv, 512, 0, stream>>>(ioutbf, verbgen_w, verbgen_b, nullptr,
                                      vlog, BBd, VV, HSZd);
}
